// PropModuleDiffDen_10411000726011
// MI455X (gfx1250) — matrix operand order repaired and checked by reading, not compiled, not run on hardware
//
#include <hip/hip_runtime.h>

// ---------------------------------------------------------------------------
// Problem constants (match reference)
// ---------------------------------------------------------------------------
#define BB   2
#define NN   2048
#define RR   16384
#define NF   128
#define NHH  3
#define PSTEP 3
#define BR   (BB * RR)   // 32768 relation rows
#define BN   (BB * NN)   // 4096 node rows

typedef __attribute__((ext_vector_type(16))) _Float16 v16h;
typedef __attribute__((ext_vector_type(8)))  float    v8f;

// ---------------------------------------------------------------------------
// WMMA fragment loaders (v_wmma_f32_16x16x32_f16, contiguous k order shared by A and B)
// A 16x32 f16: lane<16 -> row=lane, halves 0..15 = K0..15
//              lane>=16 -> row=lane-16, halves 0..15 = K16..31
// ---------------------------------------------------------------------------
__device__ __forceinline__ v16h load_a_frag(const float* __restrict__ rowp, int lane) {
  const int koff = (lane & 16) ? 16 : 0;
  v16h a;
#pragma unroll
  for (int j = 0; j < 8; ++j) {
    a[j]     = (_Float16)rowp[koff + j];
    a[8 + j] = (_Float16)rowp[koff + 8 + j];
  }
  return a;
}

// B pre-packed so each lane reads one contiguous 32B chunk:
// Wp[((ktile*2 + laneHalf) * NF + col) * 16 + i]  ==  W[(ktile*32 + laneHalf*16 + i)*NF + col]
__device__ __forceinline__ v16h load_b_frag(const _Float16* __restrict__ Wp, int ktile,
                                            int col, int lane) {
  const _Float16* p = Wp + ((size_t)(ktile * 2 + (lane >> 4)) * NF + col) * 16;
  return *(const v16h*)p;
}

// One 128-wide K segment, two output tiles sharing the A fragment.
// Alternating acc0/acc1 keeps WMMAs independent -> hazard slots hide useful work.
__device__ __forceinline__ void gemm_seg2(const float* __restrict__ arow,
                                          const _Float16* __restrict__ Wp,
                                          int ktile0, int col0, int col1, int lane,
                                          v8f& acc0, v8f& acc1) {
#pragma unroll
  for (int t = 0; t < 4; ++t) {   // 4 x K32 = one 128-wide segment
    v16h a  = load_a_frag(arow + t * 32, lane);
    v16h b0 = load_b_frag(Wp, ktile0 + t, col0, lane);
    v16h b1 = load_b_frag(Wp, ktile0 + t, col1, lane);
    acc0 = __builtin_amdgcn_wmma_f32_16x16x32_f16(false, a, false, b0,
                                                  (short)0, acc0, false, false);
    acc1 = __builtin_amdgcn_wmma_f32_16x16x32_f16(false, a, false, b1,
                                                  (short)0, acc1, false, false);
  }
}

__device__ __forceinline__ const float* seg_row(const float* __restrict__ base,
                                                const int* __restrict__ idx, int grow) {
  const int r = idx ? idx[grow] : grow;
  return base + (size_t)r * NF;
}

// ---------------------------------------------------------------------------
// Generic fused GEMM: OUT = relu( concat(seg0,seg1,seg2) @ W + bias [+ resid] )
// Rows gathered per-segment via optional index arrays. Optional scatter-add
// epilogue (Rr^T aggregation) via f32 global atomics.
// Block: 256 threads = 8 waves; block covers 32 rows x 128 cols.
// Wave w: row-half (w>>2), column tiles 2*(w&3) and 2*(w&3)+1 (two accumulators).
// ---------------------------------------------------------------------------
__global__ __launch_bounds__(256) void gemm_relu_wmma(
    const float* __restrict__ s0, const int* __restrict__ i0,
    const float* __restrict__ s1, const int* __restrict__ i1,
    const float* __restrict__ s2, const int* __restrict__ i2,
    int nseg,
    const _Float16* __restrict__ Wp, const float* __restrict__ bias,
    const float* __restrict__ resid,
    float* __restrict__ out,
    float* __restrict__ scat_out, const int* __restrict__ scat_idx) {
  const int lane = threadIdx.x & 31;
  const int wave = threadIdx.x >> 5;           // 0..7
  const int m16  = lane & 15;
  const int row0 = blockIdx.x * 32 + (wave >> 2) * 16;
  const int col0 = (wave & 3) * 32 + m16;      // tile 0 column
  const int col1 = col0 + 16;                  // tile 1 column
  const int grow = row0 + m16;                 // row whose A data this lane feeds

  v8f acc0 = {}, acc1 = {};
  gemm_seg2(seg_row(s0, i0, grow), Wp, 0, col0, col1, lane, acc0, acc1);
  if (nseg >= 2) gemm_seg2(seg_row(s1, i1, grow), Wp, 4, col0, col1, lane, acc0, acc1);
  if (nseg >= 3) gemm_seg2(seg_row(s2, i2, grow), Wp, 8, col0, col1, lane, acc0, acc1);

  // C/D layout: VGPR i -> row = i + 8*(lane>=16), col = lane%16 (+ tile col)
  const int mbase = (lane >> 4) * 8;
  const float bv0 = bias[col0];
  const float bv1 = bias[col1];
#pragma unroll
  for (int i = 0; i < 8; ++i) {
    const int r = row0 + mbase + i;
    float v0 = acc0[i] + bv0;
    float v1 = acc1[i] + bv1;
    if (resid) {
      v0 += resid[(size_t)r * NF + col0];
      v1 += resid[(size_t)r * NF + col1];
    }
    v0 = fmaxf(v0, 0.0f);
    v1 = fmaxf(v1, 0.0f);
    if (scat_out) {
      const size_t dr = (size_t)scat_idx[r] * NF;
      unsafeAtomicAdd(&scat_out[dr + col0], v0);
      unsafeAtomicAdd(&scat_out[dr + col1], v1);
    } else {
      out[(size_t)r * NF + col0] = v0;
      out[(size_t)r * NF + col1] = v1;
    }
  }
}

// ---------------------------------------------------------------------------
// One-hot (B,R,N) -> flat node index b*N + n. One block per relation row.
// ---------------------------------------------------------------------------
__global__ void extract_indices(const float* __restrict__ Rr, const float* __restrict__ Rs,
                                int* __restrict__ idx_r, int* __restrict__ idx_s) {
  const int row = blockIdx.x;             // b*R + r
  const int b   = row / RR;
  const float* pr = Rr + (size_t)row * NN;
  const float* ps = Rs + (size_t)row * NN;
  for (int n = threadIdx.x; n < NN; n += blockDim.x) {
    if (pr[n] > 0.5f) idx_r[row] = b * NN + n;
    if (ps[n] > 0.5f) idx_s[row] = b * NN + n;
  }
}

// Pack a row-major f32 weight [K][NF] into WMMA B-fragment order (f16).
__global__ void pack_w(const float* __restrict__ W, _Float16* __restrict__ Wp, int total) {
  const int id = blockIdx.x * blockDim.x + threadIdx.x;
  if (id >= total) return;
  const int i = id & 15;
  const int c = (id >> 4) & (NF - 1);
  const int h = (id >> 11) & 1;
  const int t = id >> 12;
  const int k = t * 32 + h * 16 + i;
  Wp[id] = (_Float16)W[(size_t)k * NF + c];
}

__global__ void zero_f32(float* __restrict__ p, int n) {
  const int i = blockIdx.x * blockDim.x + threadIdx.x;
  if (i < n) p[i] = 0.0f;
}

// particle encoder layer0: K = 3*NH + 1 = 10 (scalar fp32)
__global__ void pe0_kernel(const float* __restrict__ s_delta, const float* __restrict__ a_cur,
                           const float* __restrict__ w0, const float* __restrict__ b0,
                           float* __restrict__ out) {
  const int row = blockIdx.x;       // b*N + n
  const int col = threadIdx.x;      // 0..127
  const int b = row / NN, n = row % NN;
  float acc = b0[col];
#pragma unroll
  for (int h = 0; h < NHH; ++h)
#pragma unroll
    for (int d = 0; d < 3; ++d) {
      const float x = s_delta[(((size_t)b * NHH + h) * NN + n) * 3 + d];
      acc += x * w0[(h * 3 + d) * NF + col];
    }
  acc += a_cur[row] * w0[(3 * NHH) * NF + col];
  out[(size_t)row * NF + col] = fmaxf(acc, 0.0f);
}

// relation encoder layer0: rel_in = [a_r, a_s, s_r - s_s, edge], K=6 (scalar fp32)
__global__ void re0_kernel(const float* __restrict__ a_cur, const float* __restrict__ s_cur,
                           const float* __restrict__ edge,
                           const int* __restrict__ idx_r, const int* __restrict__ idx_s,
                           const float* __restrict__ w0, const float* __restrict__ b0,
                           float* __restrict__ out) {
  const int row = blockIdx.x;       // b*R + r
  const int col = threadIdx.x;
  const int ir = idx_r[row], is = idx_s[row];     // flat b*N + n
  const float x0 = a_cur[ir];
  const float x1 = a_cur[is];
  const float d0 = s_cur[ir * 3 + 0] - s_cur[is * 3 + 0];
  const float d1 = s_cur[ir * 3 + 1] - s_cur[is * 3 + 1];
  const float d2 = s_cur[ir * 3 + 2] - s_cur[is * 3 + 2];
  const float x5 = edge[row];
  float acc = b0[col]
            + x0 * w0[0 * NF + col] + x1 * w0[1 * NF + col]
            + d0 * w0[2 * NF + col] + d1 * w0[3 * NF + col] + d2 * w0[4 * NF + col]
            + x5 * w0[5 * NF + col];
  out[(size_t)row * NF + col] = fmaxf(acc, 0.0f);
}

// final predictor layer1: K=128 -> 3, plus s_cur residual (scalar fp32)
__global__ void pr1_kernel(const float* __restrict__ h, const float* __restrict__ w1,
                           const float* __restrict__ b1, const float* __restrict__ s_cur,
                           float* __restrict__ out, int nrows) {
  const int row = blockIdx.x * blockDim.x + threadIdx.x;
  if (row >= nrows) return;
  float a0 = b1[0], a1 = b1[1], a2 = b1[2];
  const float* hr = h + (size_t)row * NF;
#pragma unroll 4
  for (int k = 0; k < NF; ++k) {
    const float x = hr[k];
    a0 += x * w1[k * 3 + 0];
    a1 += x * w1[k * 3 + 1];
    a2 += x * w1[k * 3 + 2];
  }
  out[row * 3 + 0] = a0 + s_cur[row * 3 + 0];
  out[row * 3 + 1] = a1 + s_cur[row * 3 + 1];
  out[row * 3 + 2] = a2 + s_cur[row * 3 + 2];
}

// ---------------------------------------------------------------------------
extern "C" void kernel_launch(void* const* d_in, const int* in_sizes, int n_in,
                              void* d_out, int out_size, void* d_ws, size_t ws_size,
                              hipStream_t stream) {
  const float* a_cur   = (const float*)d_in[0];
  const float* s_cur   = (const float*)d_in[1];
  const float* s_delta = (const float*)d_in[2];
  const float* Rr      = (const float*)d_in[3];
  const float* Rs      = (const float*)d_in[4];
  const float* edge    = (const float*)d_in[5];
  const float* pe_w0 = (const float*)d_in[6];  const float* pe_b0 = (const float*)d_in[7];
  const float* pe_w1 = (const float*)d_in[8];  const float* pe_b1 = (const float*)d_in[9];
  const float* re_w0 = (const float*)d_in[10]; const float* re_b0 = (const float*)d_in[11];
  const float* re_w1 = (const float*)d_in[12]; const float* re_b1 = (const float*)d_in[13];
  const float* re_w2 = (const float*)d_in[14]; const float* re_b2 = (const float*)d_in[15];
  const float* rp_w  = (const float*)d_in[16]; const float* rp_b  = (const float*)d_in[17];
  const float* pp_w  = (const float*)d_in[18]; const float* pp_b  = (const float*)d_in[19];
  const float* pr_w0 = (const float*)d_in[20]; const float* pr_b0 = (const float*)d_in[21];
  const float* pr_w1 = (const float*)d_in[22]; const float* pr_b1 = (const float*)d_in[23];
  float* out = (float*)d_out;
  (void)in_sizes; (void)n_in; (void)out_size; (void)ws_size;

  // ---- workspace bump allocator ----
  char* ws = (char*)d_ws;
  size_t off = 0;
  auto bump = [&](size_t bytes) -> void* {
    void* p = ws + off;
    off += (bytes + 255) & ~(size_t)255;
    return p;
  };
  int*   idx_r   = (int*)  bump((size_t)BR * 4);
  int*   idx_s   = (int*)  bump((size_t)BR * 4);
  float* hidden0 = (float*)bump((size_t)BN * NF * 4);
  float* particle= (float*)bump((size_t)BN * NF * 4);
  float* effect  = (float*)bump((size_t)BN * NF * 4);
  float* agg     = (float*)bump((size_t)BN * NF * 4);
  float* pred_h  = (float*)bump((size_t)BN * NF * 4);
  float* relA    = (float*)bump((size_t)BR * NF * 4);
  float* relB    = (float*)bump((size_t)BR * NF * 4);
  _Float16* pe_w1h = (_Float16*)bump((size_t)128 * NF * 2);
  _Float16* re_w1h = (_Float16*)bump((size_t)128 * NF * 2);
  _Float16* re_w2h = (_Float16*)bump((size_t)128 * NF * 2);
  _Float16* rp_wh  = (_Float16*)bump((size_t)384 * NF * 2);
  _Float16* pp_wh  = (_Float16*)bump((size_t)256 * NF * 2);
  _Float16* pr_w0h = (_Float16*)bump((size_t)128 * NF * 2);

  auto packW = [&](const float* W, _Float16* Wp, int K) {
    const int n = K * NF;
    pack_w<<<dim3((n + 255) / 256), dim3(256), 0, stream>>>(W, Wp, n);
  };

  // ---- stage 0: weight packing + one-hot -> index extraction ----
  packW(pe_w1, pe_w1h, 128);
  packW(re_w1, re_w1h, 128);
  packW(re_w2, re_w2h, 128);
  packW(rp_w,  rp_wh,  384);
  packW(pp_w,  pp_wh,  256);
  packW(pr_w0, pr_w0h, 128);
  extract_indices<<<dim3(BR), dim3(256), 0, stream>>>(Rr, Rs, idx_r, idx_s);

  const int* NOIDX = nullptr;

  // ---- particle encoder ----
  pe0_kernel<<<dim3(BN), dim3(NF), 0, stream>>>(s_delta, a_cur, pe_w0, pe_b0, hidden0);
  gemm_relu_wmma<<<dim3(BN / 32), dim3(256), 0, stream>>>(
      hidden0, NOIDX, nullptr, NOIDX, nullptr, NOIDX, 1,
      pe_w1h, pe_b1, nullptr, particle, nullptr, NOIDX);

  // ---- relation encoder ----
  re0_kernel<<<dim3(BR), dim3(NF), 0, stream>>>(a_cur, s_cur, edge, idx_r, idx_s,
                                                re_w0, re_b0, relA);
  gemm_relu_wmma<<<dim3(BR / 32), dim3(256), 0, stream>>>(
      relA, NOIDX, nullptr, NOIDX, nullptr, NOIDX, 1,
      re_w1h, re_b1, nullptr, relB, nullptr, NOIDX);
  gemm_relu_wmma<<<dim3(BR / 32), dim3(256), 0, stream>>>(
      relB, NOIDX, nullptr, NOIDX, nullptr, NOIDX, 1,
      re_w2h, re_b2, nullptr, relA, nullptr, NOIDX);   // relA = relation_encode

  // ---- effect = particle_encode ----
  hipMemcpyAsync(effect, particle, (size_t)BN * NF * 4, hipMemcpyDeviceToDevice, stream);

  // ---- propagation steps ----
  for (int step = 0; step < PSTEP; ++step) {
    zero_f32<<<dim3((BN * NF + 255) / 256), dim3(256), 0, stream>>>(agg, BN * NF);
    // e_rel = relu([relation_encode, effect[idx_r], effect[idx_s]] @ rp_w + rp_b)
    // fused scatter: agg[idx_r[r]] += e_rel[r]   (== Rr^T @ e_rel)
    gemm_relu_wmma<<<dim3(BR / 32), dim3(256), 0, stream>>>(
        relA, NOIDX, effect, idx_r, effect, idx_s, 3,
        rp_wh, rp_b, nullptr, nullptr, agg, idx_r);
    // effect = relu([particle, agg] @ pp_w + pp_b + effect)
    gemm_relu_wmma<<<dim3(BN / 32), dim3(256), 0, stream>>>(
        particle, NOIDX, agg, NOIDX, nullptr, NOIDX, 2,
        pp_wh, pp_b, effect, effect, nullptr, NOIDX);
  }

  // ---- predictor ----
  gemm_relu_wmma<<<dim3(BN / 32), dim3(256), 0, stream>>>(
      effect, NOIDX, nullptr, NOIDX, nullptr, NOIDX, 1,
      pr_w0h, pr_b0, nullptr, pred_h, nullptr, NOIDX);
  pr1_kernel<<<dim3((BN + 127) / 128), dim3(128), 0, stream>>>(
      pred_h, pr_w1, pr_b1, s_cur, out, BN);
}
